// NodeFromEdgeAttention_34187939676266
// MI455X (gfx1250) — matrix operand order repaired and checked by reading, not compiled, not run on hardware
//
#include <hip/hip_runtime.h>

// NodeFromEdgeAttention for MI455X (gfx1250, wave32).
// B=16, N=128, D_Z=128, D_S=256, H=8, HD=32.
//
// Algebraic restructuring: per (b,i) the query is a single vector per head, so
//   scores[h,j] = zn[j,:] . (q[h,:] @ Wk_h)          (qk precomputed, tiny)
//   out[h,:]    = (attn[h,:] @ zn) @ Wv_h^T + bv_h   (az computed first)
// bk is softmax-invariant and dropped; bv passes through softmax unchanged.
// This turns 2x17 GFLOP of K/V projection into ~1.3 GFLOP and leaves the op
// bound by the single 134 MB read of z (~6 us at 23.3 TB/s).
// The two remaining per-tile GEMMs use v_wmma_f32_16x16x32_bf16.

#define NB  16
#define NN  128
#define DZ  128
#define DS  256
#define NH  8
#define HD  32
#define ZLD 136     // bf16 LDS row stride (padded)
#define EPSL 1e-5f

typedef __bf16 bf16;
typedef bf16  v16bf __attribute__((ext_vector_type(16)));
typedef float v8f   __attribute__((ext_vector_type(8)));

__device__ __forceinline__ float waveSum(float v) {
#pragma unroll
  for (int m = 16; m >= 1; m >>= 1) v += __shfl_xor(v, m, 32);
  return v;
}
__device__ __forceinline__ float waveMax(float v) {
#pragma unroll
  for (int m = 16; m >= 1; m >>= 1) v = fmaxf(v, __shfl_xor(v, m, 32));
  return v;
}

// ---------------------------------------------------------------------------
// Kernel A: sn = LN(s_query); q = sn@wq^T + bq; qk[h,e] = sum_d q[h,d]*wk[h*32+d,e]
// One block per (b,i). Output qk: [2048][8*128] f32 in workspace.
// ---------------------------------------------------------------------------
__global__ __launch_bounds__(256)
void nfea_qk_kernel(const float* __restrict__ s_query,
                    const float* __restrict__ wq, const float* __restrict__ bq,
                    const float* __restrict__ wk,
                    const float* __restrict__ gs, const float* __restrict__ bs,
                    float* __restrict__ qk_out) {
  __shared__ float sn[DS];
  __shared__ float qv[DS];
  __shared__ float red[8];
  const int bi = blockIdx.x;
  const int t = threadIdx.x, lane = t & 31, wave = t >> 5;

  float x = s_query[(size_t)bi * DS + t];
  float s = waveSum(x);
  if (lane == 0) red[wave] = s;
  __syncthreads();
  float tot = 0.f;
#pragma unroll
  for (int i = 0; i < 8; i++) tot += red[i];
  const float mean = tot * (1.0f / DS);
  __syncthreads();
  const float d = x - mean;
  float s2 = waveSum(d * d);
  if (lane == 0) red[wave] = s2;
  __syncthreads();
  float tv = 0.f;
#pragma unroll
  for (int i = 0; i < 8; i++) tv += red[i];
  const float rs = rsqrtf(tv * (1.0f / DS) + EPSL);
  sn[t] = d * rs * gs[t] + bs[t];
  __syncthreads();

  // q[c] = bq[c] + sn . wq[c,:]
  {
    float acc = bq[t];
    const float* wr = wq + (size_t)t * DS;
#pragma unroll 4
    for (int k = 0; k < DS; k++) acc += sn[k] * wr[k];
    qv[t] = acc;
  }
  __syncthreads();

  // qk[h][e] = sum_{d2<32} q[h*32+d2] * wk[(h*32+d2)*128 + e]
  for (int idx = t; idx < NH * DZ; idx += 256) {
    const int h = idx >> 7, e = idx & 127;
    float a = 0.f;
    const float* qh  = qv + h * HD;
    const float* wkh = wk + (size_t)(h * HD) * DZ + e;
#pragma unroll 8
    for (int d2 = 0; d2 < HD; d2++) a += qh[d2] * wkh[(size_t)d2 * DZ];
    qk_out[(size_t)bi * (NH * DZ) + idx] = a;
  }
}

// ---------------------------------------------------------------------------
// Kernel B: fused LN(z) -> scores (WMMA) -> softmax -> az (WMMA) -> az@Wv^T+bv
// One block per (b,i), 8 waves.
// ---------------------------------------------------------------------------
__global__ __launch_bounds__(256)
void nfea_attn_kernel(const float* __restrict__ z,
                      const unsigned char* __restrict__ mask,
                      const float* __restrict__ gz, const float* __restrict__ bz,
                      const float* __restrict__ qk_in,
                      const float* __restrict__ wv, const float* __restrict__ bv,
                      float* __restrict__ pre_out) {
  __shared__ bf16  zn[NN * ZLD];    // 34816 B  layernormed z tile (bf16)
  __shared__ bf16  qkb[16 * DZ];    //  4096 B  qk, heads padded to 16
  __shared__ float sc[16 * NN];     //  8192 B  scores[h][j]
  __shared__ bf16  at[16 * NN];     //  4096 B  attn[h][j] (rows 8..15 zero)
  __shared__ float az[16 * DZ];     //  8192 B  attn @ zn

  const int bi = blockIdx.x;
  const int b  = bi >> 7;
  const int t = threadIdx.x, lane = t & 31, wave = t >> 5;

  // ---- Phase 1: per-row layernorm of z[b,i,:,:] -> bf16 LDS ----
  const float* zbase = z + (size_t)bi * (NN * DZ);
  const float4 g4  = reinterpret_cast<const float4*>(gz)[lane];
  const float4 bb4 = reinterpret_cast<const float4*>(bz)[lane];
#pragma unroll 2
  for (int k = 0; k < 16; k++) {
    const int r = wave * 16 + k;
    const float4 xv = reinterpret_cast<const float4*>(zbase + (size_t)r * DZ)[lane];
    float s = waveSum(xv.x + xv.y + xv.z + xv.w);
    const float mean = s * (1.0f / DZ);
    const float d0 = xv.x - mean, d1 = xv.y - mean, d2 = xv.z - mean, d3 = xv.w - mean;
    const float ss = waveSum(d0 * d0 + d1 * d1 + d2 * d2 + d3 * d3);
    const float rs = rsqrtf(ss * (1.0f / DZ) + EPSL);
    bf16* zr = zn + r * ZLD + lane * 4;
    zr[0] = (bf16)(d0 * rs * g4.x + bb4.x);
    zr[1] = (bf16)(d1 * rs * g4.y + bb4.y);
    zr[2] = (bf16)(d2 * rs * g4.z + bb4.z);
    zr[3] = (bf16)(d3 * rs * g4.w + bb4.w);
  }
  // ---- Phase 2: qk -> bf16 LDS (pad heads 8..15 with zeros) ----
  for (int idx = t; idx < 16 * DZ; idx += 256) {
    const int h = idx >> 7;
    const float v = (h < NH) ? qk_in[(size_t)bi * (NH * DZ) + idx] : 0.f;
    qkb[idx] = (bf16)v;
  }
  __syncthreads();

  // ---- Phase 3: scores' [j, h] = zn @ qk^T  (M=128 j, N=16 h, K=128 e) ----
  {
    const int jbase = wave * 16;
    const int arow  = jbase + (lane & 15);
    const int akb   = (lane < 16) ? 0 : 8;
    const int bcol  = lane & 15;
    const int bkb   = (lane < 16) ? 0 : 8;
    v8f c = {0.f, 0.f, 0.f, 0.f, 0.f, 0.f, 0.f, 0.f};
#pragma unroll
    for (int kk = 0; kk < 4; kk++) {
      v16bf a, bm;
#pragma unroll
      for (int v = 0; v < 8; v++) {        // 16-bit A 16x32 layout
        const int kp = (v < 4) ? (akb + 2 * v) : (16 + akb + 2 * (v - 4));
        const int e = kk * 32 + kp;
        a[2 * v]     = zn[arow * ZLD + e];
        a[2 * v + 1] = zn[arow * ZLD + e + 1];
      }
#pragma unroll
      for (int v = 0; v < 8; v++) {        // 16-bit B 32x16 layout
        const int e = kk * 32 + ((v < 4) ? (bkb + 2 * v) : (16 + bkb + 2 * (v - 4)));
        bm[2 * v]     = qkb[bcol * DZ + e];
        bm[2 * v + 1] = qkb[bcol * DZ + e + 1];
      }
      c = __builtin_amdgcn_wmma_f32_16x16x32_bf16(false, a, false, bm,
                                                  (short)0, c, false, false);
    }
    const float scl = 0.1767766952966369f;  // 1/sqrt(32)
    const int h = lane & 15;
    const int rb = (lane < 16) ? 0 : 8;
#pragma unroll
    for (int v = 0; v < 8; v++) sc[h * NN + jbase + rb + v] = c[v] * scl;
  }
  __syncthreads();

  // ---- Phase 4: masked softmax over j; head h = wave; zero padded rows ----
  {
    const int h = wave;
    const unsigned char* mrow = mask + (size_t)b * NN;
    float vals[4];
#pragma unroll
    for (int q = 0; q < 4; q++) {
      const int j = lane * 4 + q;
      const float v = sc[h * NN + j];
      vals[q] = mrow[j] ? v : -1e30f;
    }
    float mx = fmaxf(fmaxf(vals[0], vals[1]), fmaxf(vals[2], vals[3]));
    mx = waveMax(mx);
    float sum = 0.f;
#pragma unroll
    for (int q = 0; q < 4; q++) { vals[q] = __expf(vals[q] - mx); sum += vals[q]; }
    sum = waveSum(sum);
    const float inv = 1.0f / sum;
#pragma unroll
    for (int q = 0; q < 4; q++) at[h * NN + lane * 4 + q] = (bf16)(vals[q] * inv);
    const bf16 z0 = (bf16)0.f;
#pragma unroll
    for (int q = 0; q < 4; q++) at[(8 + wave) * NN + lane * 4 + q] = z0;
  }
  __syncthreads();

  // ---- Phase 5: az[h,e] = attn @ zn  (M=16 h, N=128 e, K=128 j) ----
  {
    const int ebase = wave * 16;
    const int arow  = lane & 15;          // head row
    const int akb   = (lane < 16) ? 0 : 8;
    const int ecol  = ebase + (lane & 15);
    const int bkb   = (lane < 16) ? 0 : 8;
    v8f c = {0.f, 0.f, 0.f, 0.f, 0.f, 0.f, 0.f, 0.f};
#pragma unroll
    for (int kk = 0; kk < 4; kk++) {
      v16bf a, bm;
#pragma unroll
      for (int v = 0; v < 8; v++) {
        const int kp = (v < 4) ? (akb + 2 * v) : (16 + akb + 2 * (v - 4));
        const int j = kk * 32 + kp;
        a[2 * v]     = at[arow * NN + j];
        a[2 * v + 1] = at[arow * NN + j + 1];
      }
#pragma unroll
      for (int v = 0; v < 8; v++) {
        const int j = kk * 32 + ((v < 4) ? (bkb + 2 * v) : (16 + bkb + 2 * (v - 4)));
        bm[2 * v]     = zn[j * ZLD + ecol];
        bm[2 * v + 1] = zn[(j + 1) * ZLD + ecol];
      }
      c = __builtin_amdgcn_wmma_f32_16x16x32_bf16(false, a, false, bm,
                                                  (short)0, c, false, false);
    }
    const int rb = (lane < 16) ? 0 : 8;
#pragma unroll
    for (int v = 0; v < 8; v++) az[(rb + v) * DZ + ecol] = c[v];
  }
  __syncthreads();

  // ---- Phase 6: out[c] = bv[c] + az[h,:] . wv[c,:]  (c = h*32+d) ----
  {
    const int c = t;
    const int h = c >> 5;
    float acc = bv[c];
    const float4* wr = reinterpret_cast<const float4*>(wv + (size_t)c * DZ);
    const float4* ar = reinterpret_cast<const float4*>(az + h * DZ);
#pragma unroll 8
    for (int e4 = 0; e4 < DZ / 4; e4++) {
      const float4 w4 = wr[e4];
      const float4 a4 = ar[e4];
      acc += w4.x * a4.x + w4.y * a4.y + w4.z * a4.z + w4.w * a4.w;
    }
    pre_out[(size_t)bi * DS + c] = acc;
  }
}

// ---------------------------------------------------------------------------
// Kernel C: final projection  out = pre @ wo^T + bo
// ---------------------------------------------------------------------------
__global__ __launch_bounds__(256)
void nfea_oproj_kernel(const float* __restrict__ pre,
                       const float* __restrict__ wo, const float* __restrict__ bo,
                       float* __restrict__ out) {
  __shared__ float xr[DS];
  const int bi = blockIdx.x, t = threadIdx.x;
  xr[t] = pre[(size_t)bi * DS + t];
  __syncthreads();
  float acc = bo[t];
  const float4* wr = reinterpret_cast<const float4*>(wo + (size_t)t * DS);
  const float4* ar = reinterpret_cast<const float4*>(xr);
#pragma unroll 8
  for (int k = 0; k < DS / 4; k++) {
    const float4 w4 = wr[k];
    const float4 a4 = ar[k];
    acc += w4.x * a4.x + w4.y * a4.y + w4.z * a4.z + w4.w * a4.w;
  }
  out[(size_t)bi * DS + t] = acc;
}

extern "C" void kernel_launch(void* const* d_in, const int* in_sizes, int n_in,
                              void* d_out, int out_size, void* d_ws, size_t ws_size,
                              hipStream_t stream) {
  const float* z  = (const float*)d_in[0];
  const float* sq = (const float*)d_in[1];
  const unsigned char* mask = (const unsigned char*)d_in[2];
  const float* wq = (const float*)d_in[3];
  const float* bq = (const float*)d_in[4];
  const float* wk = (const float*)d_in[5];
  // d_in[6] = bk: softmax-invariant (adds a per-(b,i,h) constant to all
  // scores) -> intentionally unused.
  const float* wv = (const float*)d_in[7];
  const float* bv = (const float*)d_in[8];
  const float* wo = (const float*)d_in[9];
  const float* bo = (const float*)d_in[10];
  const float* gz = (const float*)d_in[11];
  const float* bz = (const float*)d_in[12];
  const float* gs = (const float*)d_in[13];
  const float* bs = (const float*)d_in[14];

  const int NBI = NB * NN;                        // 2048 (b,i) pairs
  float* qk_ws  = (float*)d_ws;                   // [2048][1024] f32 = 8 MB
  float* pre_ws = qk_ws + (size_t)NBI * (NH * DZ);// [2048][256]  f32 = 2 MB

  dim3 blk(256);
  nfea_qk_kernel<<<NBI, blk, 0, stream>>>(sq, wq, bq, wk, gs, bs, qk_ws);
  nfea_attn_kernel<<<NBI, blk, 0, stream>>>(z, mask, gz, bz, qk_ws, wv, bv, pre_ws);
  nfea_oproj_kernel<<<NBI, blk, 0, stream>>>(pre_ws, wo, bo, (float*)d_out);
}
